// ConvGeodesic_30348238913828
// MI455X (gfx1250) — matrix operand order repaired and checked by reading, not compiled, not run on hardware
//
#include <hip/hip_runtime.h>
#include <hip/hip_bf16.h>

typedef __bf16 v16bf __attribute__((ext_vector_type(16)));
typedef float  v8f   __attribute__((ext_vector_type(8)));

// Problem dims (B, R, A, K, O, I) = (50000, 5, 8, 2, 32, 32)
constexpr int Bn = 50000, Rn = 5, An = 8, Kn = 2, On = 32, In = 32;
constexpr int KDIM = Rn * An * In;      // 1280  (GEMM K: (r,a,i))
constexpr int NDIM = An * On;           // 256   (GEMM N: (c,o))
constexpr int KT   = KDIM / 32;         // 40 K-tiles of 32
constexpr int NT   = NDIM / 16;         // 16 N-tiles of 16
constexpr int MT   = 16;                // rows (b) per block
constexpr int ROWU = KDIM + 8;          // ushorts per LDS pullback row (16B pad -> bank-conflict-free)
constexpr int WELEMS = KT * NT * 32 * 16; // 327680 packed bf16 per weight copy

__device__ __forceinline__ unsigned short f2bf(float x) {
  union { float f; unsigned u; } v; v.f = x;
  unsigned r = v.u + 0x7FFFu + ((v.u >> 16) & 1u);   // round-to-nearest-even
  return (unsigned short)(r >> 16);
}
__device__ __forceinline__ float bf2f(unsigned short h) {
  union { unsigned u; float f; } v; v.u = ((unsigned)h) << 16; return v.f;
}

union Frag { uint4 q[2]; v16bf v; };

// ---------------------------------------------------------------------------
// Prep: fold k, apply rotation (c+a)%8, split fp32 -> bf16 hi/lo, and store in
// this kernel's contiguous-k B-fragment order: [kt][nt][lane][elem], lane<16: N=lane,K=elem;
// lane>=16: N=lane-16, K=16+elem.  Each lane's 16 bf16 are 32B contiguous.
// ---------------------------------------------------------------------------
__global__ void pack_weights(const float* __restrict__ kern,
                             unsigned short* __restrict__ whi,
                             unsigned short* __restrict__ wlo) {
  int idx = blockIdx.x * 256 + threadIdx.x;
  if (idx >= WELEMS) return;
  int e    = idx & 15;
  int lane = (idx >> 4) & 31;
  int nt   = (idx >> 9) & 15;
  int kt   = idx >> 13;
  int kg = kt * 32 + ((lane >> 4) ? 16 : 0) + e;   // global K index (r,a,i)
  int n  = nt * 16 + (lane & 15);                  // global N index (c,o)
  int r = kg >> 8;            // kg / 256
  int a = (kg >> 5) & 7;
  int i = kg & 31;
  int c = n >> 5;
  int o = n & 31;
  int cp = (c + a) & 7;       // rotation
  // kernel[r][cp][k][o][i], k-stride = O*I = 1024
  const float* p = kern + ((((r * 8 + cp) * 2) * 32 + o) * 32 + i);
  float w = p[0] + p[1024];   // fold Sum_k
  unsigned short hi = f2bf(w);
  whi[idx] = hi;
  wlo[idx] = f2bf(w - bf2f(hi));
}

// ---------------------------------------------------------------------------
// Main: per block, 16 rows of b.  pullback -> LDS (bf16 hi/lo), split-bf16
// WMMA GEMM (each wave owns channel c = wave), then relu/norm/argmax/gather.
// ---------------------------------------------------------------------------
__global__ __launch_bounds__(256) void conv_main(
    const float* __restrict__ signal, const float* __restrict__ bary,
    const unsigned short* __restrict__ whi, const unsigned short* __restrict__ wlo,
    float* __restrict__ out) {
  extern __shared__ __align__(16) char smem[];
  unsigned short* sPhi = (unsigned short*)smem;                       // MT*ROWU
  unsigned short* sPlo = (unsigned short*)(smem + MT * ROWU * 2);     // MT*ROWU
  float* sAct  = (float*)(smem + 2 * MT * ROWU * 2);                  // MT*NDIM
  float* sNorm = (float*)(smem + 2 * MT * ROWU * 2 + MT * NDIM * 4);  // MT*8
  int*   sBest = (int*)  (smem + 2 * MT * ROWU * 2 + MT * NDIM * 4 + MT * 8 * 4);

  const int  tid = threadIdx.x;
  const long b0  = (long)blockIdx.x * MT;

  // ---- Phase 1: pullback[m, (r,a,i)] = Sum_t bary*signal, as bf16 hi/lo ----
  for (int m = 0; m < MT; ++m) {
    const float* sig = signal + (b0 + m) * (long)(Rn * An * 3 * In);
    const float* bry = bary   + (b0 + m) * (long)(Rn * An * 3);
    for (int j = 0; j < KDIM / 256; ++j) {
      int k  = j * 256 + tid;            // 0..1279
      int ra = k >> 5, i = k & 31;
      const float* sp = sig + ra * 96 + i;             // t-stride = 32
      float p = bry[ra * 3 + 0] * sp[0]
              + bry[ra * 3 + 1] * sp[32]
              + bry[ra * 3 + 2] * sp[64];
      unsigned short hi = f2bf(p);
      sPhi[m * ROWU + k] = hi;
      sPlo[m * ROWU + k] = f2bf(p - bf2f(hi));
    }
  }
  __syncthreads();

  // ---- Phase 2: GEMM.  Wave w owns N-tiles 2w, 2w+1 (channel c = w). ----
  const int lane   = tid & 31;
  const int wave   = tid >> 5;
  const int hiHalf = lane >> 4;          // A: K-half select; C: M-half select
  const int col    = lane & 15;          // A: row M; B/C: column N
  const int nt0 = wave * 2, nt1 = wave * 2 + 1;
  v8f acc0 = {0.f, 0.f, 0.f, 0.f, 0.f, 0.f, 0.f, 0.f};
  v8f acc1 = acc0;

  for (int kt = 0; kt < KT; ++kt) {
    Frag ahi, alo, b0h, b0l, b1h, b1l;
    // A fragment (contiguous k, same order as packed B): lane holds row=col, halves by hiHalf
    const unsigned short* pa = &sPhi[col * ROWU + kt * 32 + hiHalf * 16];
    ahi.q[0] = *(const uint4*)pa;       ahi.q[1] = *(const uint4*)(pa + 8);
    const unsigned short* pl = &sPlo[col * ROWU + kt * 32 + hiHalf * 16];
    alo.q[0] = *(const uint4*)pl;       alo.q[1] = *(const uint4*)(pl + 8);
    // B fragments: pre-swizzled, 32B contiguous per lane
    const unsigned short* pb0 = whi + (((kt * NT + nt0) * 32 + lane) << 4);
    const unsigned short* pb1 = whi + (((kt * NT + nt1) * 32 + lane) << 4);
    const unsigned short* qb0 = wlo + (((kt * NT + nt0) * 32 + lane) << 4);
    const unsigned short* qb1 = wlo + (((kt * NT + nt1) * 32 + lane) << 4);
    b0h.q[0] = *(const uint4*)pb0;      b0h.q[1] = *(const uint4*)(pb0 + 8);
    b1h.q[0] = *(const uint4*)pb1;      b1h.q[1] = *(const uint4*)(pb1 + 8);
    b0l.q[0] = *(const uint4*)qb0;      b0l.q[1] = *(const uint4*)(qb0 + 8);
    b1l.q[0] = *(const uint4*)qb1;      b1l.q[1] = *(const uint4*)(qb1 + 8);

    // bf16x3 split-precision: hi*hi + hi*lo + lo*hi, fp32 accumulate
    acc0 = __builtin_amdgcn_wmma_f32_16x16x32_bf16(false, ahi.v, false, b0h.v, (short)0, acc0, false, false);
    acc0 = __builtin_amdgcn_wmma_f32_16x16x32_bf16(false, ahi.v, false, b0l.v, (short)0, acc0, false, false);
    acc0 = __builtin_amdgcn_wmma_f32_16x16x32_bf16(false, alo.v, false, b0h.v, (short)0, acc0, false, false);
    acc1 = __builtin_amdgcn_wmma_f32_16x16x32_bf16(false, ahi.v, false, b1h.v, (short)0, acc1, false, false);
    acc1 = __builtin_amdgcn_wmma_f32_16x16x32_bf16(false, ahi.v, false, b1l.v, (short)0, acc1, false, false);
    acc1 = __builtin_amdgcn_wmma_f32_16x16x32_bf16(false, alo.v, false, b1h.v, (short)0, acc1, false, false);
  }

  // ---- Phase 3: relu -> LDS act tile (C layout: VGPR v -> M = v + 8*hiHalf)
  #pragma unroll
  for (int v = 0; v < 8; ++v) {
    int m = v + hiHalf * 8;
    float x0 = acc0[v]; x0 = x0 > 0.f ? x0 : 0.f;
    float x1 = acc1[v]; x1 = x1 > 0.f ? x1 : 0.f;
    sAct[m * NDIM + nt0 * 16 + col] = x0;
    sAct[m * NDIM + nt1 * 16 + col] = x1;
  }
  __syncthreads();

  // per-(m,c) sum of squares
  if (tid < MT * An) {                   // 128 threads
    int m = tid >> 3, c = tid & 7;
    const float* p = &sAct[m * NDIM + c * 32];
    float s = 0.f;
    #pragma unroll
    for (int o = 0; o < 32; ++o) { float x = p[o]; s += x * x; }
    sNorm[m * 8 + c] = s;
  }
  __syncthreads();

  // argmax over c (first-max, matching jnp.argmax)
  if (tid < MT) {
    float bv = sNorm[tid * 8]; int bi = 0;
    for (int c = 1; c < 8; ++c) {
      float v = sNorm[tid * 8 + c];
      if (v > bv) { bv = v; bi = c; }
    }
    sBest[tid] = bi;
  }
  __syncthreads();

  // gather winning channel, coalesced store
  for (int t = tid; t < MT * On; t += 256) {
    int m = t >> 5, o = t & 31;
    out[(b0 + m) * On + o] = sAct[m * NDIM + sBest[m] * 32 + o];
  }
}

// ---------------------------------------------------------------------------
extern "C" void kernel_launch(void* const* d_in, const int* in_sizes, int n_in,
                              void* d_out, int out_size, void* d_ws, size_t ws_size,
                              hipStream_t stream) {
  (void)in_sizes; (void)n_in; (void)out_size; (void)ws_size;
  const float* signal = (const float*)d_in[0];
  const float* bary   = (const float*)d_in[1];
  const float* kern   = (const float*)d_in[2];
  float* out = (float*)d_out;

  unsigned short* whi = (unsigned short*)d_ws;        // 327680 bf16
  unsigned short* wlo = whi + WELEMS;                 // 327680 bf16 (1.31 MB total)

  pack_weights<<<WELEMS / 256, 256, 0, stream>>>(kern, whi, wlo);

  constexpr size_t SMEM = (size_t)(2 * MT * ROWU * 2)   // pullback hi/lo bf16
                        + (size_t)(MT * NDIM * 4)       // act tile
                        + (size_t)(MT * 8 * 4)          // norms
                        + (size_t)(MT * 4);             // argmax
  conv_main<<<Bn / MT, 256, SMEM, stream>>>(signal, bary, whi, wlo, out);
}
